// MultiHeadAttention_24369644438128
// MI455X (gfx1250) — compile-verified, matrix operand order checked by reading, not run on hardware
//
#include <hip/hip_runtime.h>
#include <hip/hip_bf16.h>

typedef _Float16 h16;
typedef __attribute__((ext_vector_type(4)))  _Float16 v4h;
typedef __attribute__((ext_vector_type(8)))  _Float16 v8h;
typedef __attribute__((ext_vector_type(16))) _Float16 v16h;
typedef __attribute__((ext_vector_type(8)))  float    v8f;

#define NB 32      // batch
#define TSEQ 512   // sequence length
#define CEMB 512   // embedding / hidden
#define NH 8       // heads
#define HS 64      // head size

__device__ __forceinline__ v16h pack16(v8h lo, v8h hi) {
  v16h r;
#pragma unroll
  for (int i = 0; i < 8; ++i) { r[i] = lo[i]; r[i + 8] = hi[i]; }
  return r;
}

__device__ __forceinline__ v8f wmma_f16(v16h a, v16h b, v8f c) {
  // v_wmma_f32_16x16x32_f16: D = A(16x32) * B(32x16) + C(16x16 f32)
  return __builtin_amdgcn_wmma_f32_16x16x32_f16(false, a, false, b, (short)0, c,
                                                false, false);
}

// Async global->LDS copy, 16 bytes per lane (GLOBAL_LOAD_ASYNC_TO_LDS_B128).
// IMPORTANT: take the real LDS pointer so the shared array's address escapes
// into the asm (otherwise GlobalOpt proves "never stored" and folds all LDS
// loads to undef). Low 32 bits of the generic pointer == LDS byte offset.
__device__ __forceinline__ void async_copy_b128(void* lds, const void* g) {
  unsigned ldsoff = (unsigned)(uintptr_t)lds;
  asm volatile("global_load_async_to_lds_b128 %0, %1, off"
               :: "v"(ldsoff), "v"((unsigned long long)(uintptr_t)g)
               : "memory");
}
__device__ __forceinline__ void async_wait0() {
  asm volatile("s_wait_asynccnt 0x0" ::: "memory");
}

// ---------------------------------------------------------------------------
// Kernel 0a: repack Wq/Wk/Wv/Wo (fp32 row-major [512][512]) into f16 B-operand
// tiles: WB[m][nt][kc][lane][16 halves]; lane holds column nt*16+(lane&15),
// K pattern: e<8 -> kc*32 + hi*8 + e ; e>=8 -> kc*32 + 16 + hi*8 + (e-8)
// ---------------------------------------------------------------------------
__global__ __launch_bounds__(256) void k_convert_w(
    const float* __restrict__ Wq, const float* __restrict__ Wk,
    const float* __restrict__ Wv, const float* __restrict__ Wo,
    h16* __restrict__ out) {
  int idx = blockIdx.x * 256 + threadIdx.x;       // 0 .. 4*262144-1
  int m    = idx >> 18;
  int r    = idx & 262143;
  int e    = r & 15;
  int lane = (r >> 4) & 31;
  int kc   = (r >> 9) & 15;
  int nt   = r >> 13;
  int hiL  = lane >> 4;
  int col  = nt * 16 + (lane & 15);
  int kk   = kc * 32 + ((e < 8) ? (hiL * 8 + e) : (16 + hiL * 8 + (e - 8)));
  const float* W = (m == 0) ? Wq : ((m == 1) ? Wk : ((m == 2) ? Wv : Wo));
  out[idx] = (h16)W[kk * CEMB + col];
}

// ---------------------------------------------------------------------------
// Kernel 0b: convert x fp32 -> f16 row-major (enables async b128 LDS staging).
// ---------------------------------------------------------------------------
__global__ __launch_bounds__(256) void k_convert_x(const float* __restrict__ x,
                                                   h16* __restrict__ Xh) {
  int idx = blockIdx.x * 256 + threadIdx.x;       // 1,048,576 units of 8
  const float4 f0 = *(const float4*)(x + (size_t)idx * 8);
  const float4 f1 = *(const float4*)(x + (size_t)idx * 8 + 4);
  v8h o;
  o[0] = (h16)f0.x; o[1] = (h16)f0.y; o[2] = (h16)f0.z; o[3] = (h16)f0.w;
  o[4] = (h16)f1.x; o[5] = (h16)f1.y; o[6] = (h16)f1.z; o[7] = (h16)f1.w;
  *(v8h*)(Xh + (size_t)idx * 8) = o;
}

// ---------------------------------------------------------------------------
// Kernel 1: fused QKV projection + RoPE.
// Block = one 16-row x tile (async-staged to LDS), 8 waves; each wave does
// 6 PAIRS of adjacent 16x16 output tiles (two independent WMMA chains per
// shared A fragment). Q,K stored f16 [b][h][t][d] after RoPE; V transposed
// [b][h][d][t] (B-operand layout for P*V).
// ---------------------------------------------------------------------------
__global__ __launch_bounds__(256) void k_qkv(
    const h16* __restrict__ Xh, const h16* __restrict__ WB,
    h16* __restrict__ Q, h16* __restrict__ K, h16* __restrict__ Vt) {
  __shared__ h16 xs[16 * 520];
  const int tid = threadIdx.x;
  const int t0g = blockIdx.x * 16;

  // async-stage x tile (16 x 512 halves, padded row stride 520)
#pragma unroll
  for (int i = 0; i < 4; ++i) {
    int u = tid + i * 256;          // 1024 16B units
    int row = u >> 6, c8 = u & 63;
    async_copy_b128(&xs[row * 520 + c8 * 8],
                    Xh + (size_t)(t0g + row) * CEMB + c8 * 8);
  }
  async_wait0();
  __syncthreads();

  const int wave = tid >> 5, lane = tid & 31;
  const int ln = lane & 15, hi = lane >> 4;
  const int b = t0g >> 9;
  const int tl0 = t0g & (TSEQ - 1);

  auto epilogue = [&](int m, int nt, v8f acc) {
    const int jcol = nt * 16 + ln;
    const int h = jcol >> 6, dl = jcol & 63;
    const int bh = b * NH + h;
    if (m < 2) {
      h16* dst = (m == 0) ? Q : K;
      const float freq =
          __expf(-(float)(dl >> 1) * (2.0f / 64.0f) * 9.210340371976184f);
#pragma unroll
      for (int r = 0; r < 8; ++r) {
        int M = r + hi * 8;
        float t = (float)(tl0 + M);
        float sv, cv;
        __sincosf(t * freq, &sv, &cv);
        float v = acc[r];
        float pv = __shfl_xor(v, 1, 32);   // partner dim (adjacent lane)
        float o = (dl & 1) ? (v * cv + pv * sv) : (v * cv - pv * sv);
        dst[((size_t)bh * TSEQ + (tl0 + M)) * HS + dl] = (h16)o;
      }
    } else {
#pragma unroll
      for (int r = 0; r < 8; ++r) {
        int M = r + hi * 8;
        Vt[((size_t)bh * HS + dl) * TSEQ + (tl0 + M)] = (h16)acc[r];
      }
    }
  };

  for (int jp = 0; jp < 6; ++jp) {
    int id0 = wave * 12 + jp * 2;   // (even, even+1): same matrix, adjacent nt
    int m   = id0 >> 5;             // 0=Q 1=K 2=V
    int nt0 = id0 & 31;
    const h16* wbase = WB + (size_t)m * 262144;
    v8f acc0 = {}, acc1 = {};
#pragma unroll
    for (int kc = 0; kc < 16; ++kc) {
      const h16* xr = &xs[ln * 520 + kc * 32 + hi * 8];
      v16h a = pack16(*(const v8h*)xr, *(const v8h*)(xr + 16));
      const h16* w0 = wbase + ((size_t)(nt0 * 16 + kc) * 32 + lane) * 16;
      const h16* w1 = w0 + 16 * 32 * 16;   // next n-tile
      v16h b0 = pack16(*(const v8h*)w0, *(const v8h*)(w0 + 8));
      v16h b1 = pack16(*(const v8h*)w1, *(const v8h*)(w1 + 8));
      acc0 = wmma_f16(a, b0, acc0);
      acc1 = wmma_f16(a, b1, acc1);
    }
    epilogue(m, nt0, acc0);
    epilogue(m, nt0 + 1, acc1);
  }
}

// ---------------------------------------------------------------------------
// Kernel 2: flash-style causal attention. One wave per (b,h,16-row q tile).
// Streams 32-key blocks: 4 WMMAs (S) + online softmax + 4 WMMAs (P*V).
// ---------------------------------------------------------------------------
__global__ __launch_bounds__(32) void k_attn(
    const h16* __restrict__ Q, const h16* __restrict__ K,
    const h16* __restrict__ Vt, h16* __restrict__ Y) {
  __shared__ h16 ps[16 * 40];  // P tile staging (row stride 40 halves, 16B ok)
  const int lane = threadIdx.x & 31;
  const int ln = lane & 15, hi = lane >> 4;
  const int qt = blockIdx.x & 31;
  const int h  = (blockIdx.x >> 5) & 7;
  const int b  = blockIdx.x >> 8;
  const int bh = b * NH + h;
  const int tq0 = qt * 16;

  const h16* qrow = Q + ((size_t)bh * TSEQ + tq0 + ln) * HS;
  v16h aq0 = pack16(*(const v8h*)(qrow + hi * 8),
                    *(const v8h*)(qrow + 16 + hi * 8));
  v16h aq1 = pack16(*(const v8h*)(qrow + 32 + hi * 8),
                    *(const v8h*)(qrow + 48 + hi * 8));

  v8f O0 = {}, O1 = {}, O2 = {}, O3 = {};
  float mrow[8], lrow[8];
#pragma unroll
  for (int r = 0; r < 8; ++r) { mrow[r] = -__builtin_inff(); lrow[r] = 0.0f; }

  const int nkb = ((tq0 + 15) >> 5) + 1;
  for (int kb = 0; kb < nkb; ++kb) {
    const int tk0 = kb * 32;
    // prefetch next key block's K and V lines while we compute this one
    if (kb + 1 < nkb) {
      const int tkn = tk0 + 32;
      __builtin_prefetch(K + ((size_t)bh * TSEQ + tkn + ln) * HS + hi * 8, 0, 1);
      __builtin_prefetch(Vt + ((size_t)bh * HS + ln) * TSEQ + tkn + hi * 8, 0, 1);
    }
    v8f s0 = {}, s1 = {};
    {
      const h16* kr0 = K + ((size_t)bh * TSEQ + tk0 + ln) * HS;
      s0 = wmma_f16(aq0, pack16(*(const v8h*)(kr0 + hi * 8),
                                *(const v8h*)(kr0 + 16 + hi * 8)), s0);
      s0 = wmma_f16(aq1, pack16(*(const v8h*)(kr0 + 32 + hi * 8),
                                *(const v8h*)(kr0 + 48 + hi * 8)), s0);
      const h16* kr1 = kr0 + 16 * HS;
      s1 = wmma_f16(aq0, pack16(*(const v8h*)(kr1 + hi * 8),
                                *(const v8h*)(kr1 + 16 + hi * 8)), s1);
      s1 = wmma_f16(aq1, pack16(*(const v8h*)(kr1 + 32 + hi * 8),
                                *(const v8h*)(kr1 + 48 + hi * 8)), s1);
    }
    const int tkA = tk0 + ln, tkB = tk0 + 16 + ln;
#pragma unroll
    for (int r = 0; r < 8; ++r) {
      const int tq = tq0 + r + hi * 8;
      float a0 = (tkA <= tq) ? s0[r] * 0.125f : -__builtin_inff();
      float a1 = (tkB <= tq) ? s1[r] * 0.125f : -__builtin_inff();
      float mx = fmaxf(a0, a1);
#pragma unroll
      for (int o = 1; o < 16; o <<= 1) mx = fmaxf(mx, __shfl_xor(mx, o, 32));
      float mn = fmaxf(mrow[r], mx);
      float al = __expf(mrow[r] - mn);
      float p0 = __expf(a0 - mn);
      float p1 = __expf(a1 - mn);
      float rs = p0 + p1;
#pragma unroll
      for (int o = 1; o < 16; o <<= 1) rs += __shfl_xor(rs, o, 32);
      lrow[r] = lrow[r] * al + rs;
      mrow[r] = mn;
      O0[r] *= al; O1[r] *= al; O2[r] *= al; O3[r] *= al;
      const int M = r + hi * 8;
      ps[M * 40 + ln]      = (h16)p0;
      ps[M * 40 + 16 + ln] = (h16)p1;
    }
    __syncthreads();  // D-layout -> A-layout re-stripe through LDS
    v16h ap = pack16(*(const v8h*)(&ps[ln * 40 + hi * 8]),
                     *(const v8h*)(&ps[ln * 40 + 16 + hi * 8]));
    __syncthreads();
    const h16* vb = Vt + (size_t)bh * HS * TSEQ + tk0;
    {
      const h16* vr = vb + (size_t)(0 * 16 + ln) * TSEQ;
      O0 = wmma_f16(ap, pack16(*(const v8h*)(vr + hi * 8),
                               *(const v8h*)(vr + 16 + hi * 8)), O0);
    }
    {
      const h16* vr = vb + (size_t)(1 * 16 + ln) * TSEQ;
      O1 = wmma_f16(ap, pack16(*(const v8h*)(vr + hi * 8),
                               *(const v8h*)(vr + 16 + hi * 8)), O1);
    }
    {
      const h16* vr = vb + (size_t)(2 * 16 + ln) * TSEQ;
      O2 = wmma_f16(ap, pack16(*(const v8h*)(vr + hi * 8),
                               *(const v8h*)(vr + 16 + hi * 8)), O2);
    }
    {
      const h16* vr = vb + (size_t)(3 * 16 + ln) * TSEQ;
      O3 = wmma_f16(ap, pack16(*(const v8h*)(vr + hi * 8),
                               *(const v8h*)(vr + 16 + hi * 8)), O3);
    }
  }

#pragma unroll
  for (int r = 0; r < 8; ++r) {
    const int M = r + hi * 8;
    float inv = 1.0f / lrow[r];
    size_t base = ((size_t)(b * TSEQ + tq0 + M)) * CEMB + h * HS;
    Y[base + ln]      = (h16)(O0[r] * inv);
    Y[base + 16 + ln] = (h16)(O1[r] * inv);
    Y[base + 32 + ln] = (h16)(O2[r] * inv);
    Y[base + 48 + ln] = (h16)(O3[r] * inv);
  }
}

// ---------------------------------------------------------------------------
// Kernel 3: out = Y @ Wo + bo (fp32 output). Async LDS staging + paired tiles.
// ---------------------------------------------------------------------------
__global__ __launch_bounds__(256) void k_proj(
    const h16* __restrict__ Y, const h16* __restrict__ WoB,
    const float* __restrict__ bo, float* __restrict__ out) {
  __shared__ h16 ys[16 * 520];
  const int tid = threadIdx.x;
  const int t0g = blockIdx.x * 16;
#pragma unroll
  for (int i = 0; i < 4; ++i) {
    int u = tid + i * 256;          // 1024 16B units
    int row = u >> 6, c8 = u & 63;
    async_copy_b128(&ys[row * 520 + c8 * 8],
                    Y + (size_t)(t0g + row) * CEMB + c8 * 8);
  }
  async_wait0();
  __syncthreads();
  const int wave = tid >> 5, lane = tid & 31;
  const int ln = lane & 15, hi = lane >> 4;
  for (int jp = 0; jp < 2; ++jp) {
    int nt0 = wave * 4 + jp * 2;
    v8f acc0 = {}, acc1 = {};
#pragma unroll
    for (int kc = 0; kc < 16; ++kc) {
      const h16* yr = &ys[ln * 520 + kc * 32 + hi * 8];
      v16h a = pack16(*(const v8h*)yr, *(const v8h*)(yr + 16));
      const h16* w0 = WoB + ((size_t)(nt0 * 16 + kc) * 32 + lane) * 16;
      const h16* w1 = w0 + 16 * 32 * 16;
      v16h b0 = pack16(*(const v8h*)w0, *(const v8h*)(w0 + 8));
      v16h b1 = pack16(*(const v8h*)w1, *(const v8h*)(w1 + 8));
      acc0 = wmma_f16(a, b0, acc0);
      acc1 = wmma_f16(a, b1, acc1);
    }
#pragma unroll
    for (int half = 0; half < 2; ++half) {
      int nt = nt0 + half;
      v8f acc = half ? acc1 : acc0;
      float bias = bo[nt * 16 + ln];
#pragma unroll
      for (int r = 0; r < 8; ++r) {
        int M = r + hi * 8;
        out[(size_t)(t0g + M) * CEMB + nt * 16 + ln] = acc[r] + bias;
      }
    }
  }
}

// ---------------------------------------------------------------------------
extern "C" void kernel_launch(void* const* d_in, const int* in_sizes, int n_in,
                              void* d_out, int out_size, void* d_ws,
                              size_t ws_size, hipStream_t stream) {
  (void)in_sizes; (void)n_in; (void)out_size; (void)ws_size;
  const float* x  = (const float*)d_in[0];
  const float* Wq = (const float*)d_in[1];
  const float* Wk = (const float*)d_in[2];
  const float* Wv = (const float*)d_in[3];
  const float* Wo = (const float*)d_in[4];
  const float* bo = (const float*)d_in[5];
  float* out = (float*)d_out;

  // workspace (halves): 4 repacked weights + Xh + Q + K + Vt + Y (~82 MB)
  h16* ws = (h16*)d_ws;
  h16* WB = ws;                        // 4 * 262144 halves
  h16* Xh = ws + (size_t)4 * 262144;   // 8388608 halves each from here on
  h16* Q  = Xh + (size_t)8388608;
  h16* K  = Q  + (size_t)8388608;
  h16* Vt = K  + (size_t)8388608;
  h16* Y  = Vt + (size_t)8388608;

  k_convert_w<<<4096, 256, 0, stream>>>(Wq, Wk, Wv, Wo, WB);
  k_convert_x<<<4096, 256, 0, stream>>>(x, Xh);
  k_qkv<<<(NB * TSEQ) / 16, 256, 0, stream>>>(Xh, WB, Q, K, Vt);
  k_attn<<<NB * NH * (TSEQ / 16), 32, 0, stream>>>(Q, K, Vt, Y);
  k_proj<<<(NB * TSEQ) / 16, 256, 0, stream>>>(Y, WB + (size_t)3 * 262144, bo, out);
}
